// MoEResNetBKLayer_46153718563304
// MI455X (gfx1250) — matrix operand order repaired and checked by reading, not compiled, not run on hardware
//
#include <hip/hip_runtime.h>
#include <math.h>

typedef __attribute__((ext_vector_type(16))) _Float16 v16h;
typedef __attribute__((ext_vector_type(8)))  _Float16 v8h;
typedef __attribute__((ext_vector_type(8)))  float    v8f;

#define BB 2
#define NN 2048
#define DD 768
#define EE 8
#define FF 3072
#define TOK (BB*NN)      // 4096
#define PAIRS (TOK*2)    // 8192
#define TM 32            // tokens per block (two 16-row WMMA M-groups)
#define XPAD 776         // 768 + 8 halves (4-bank stagger)
#define HPAD 3080        // 3072 + 8 halves

// ---------------- complex helpers ----------------
__device__ __forceinline__ float2 cmul(float2 a, float2 b) {
    return make_float2(a.x*b.x - a.y*b.y, a.x*b.y + a.y*b.x);
}
__device__ __forceinline__ float2 cadd(float2 a, float2 b) { return make_float2(a.x+b.x, a.y+b.y); }
__device__ __forceinline__ float2 csub(float2 a, float2 b) { return make_float2(a.x-b.x, a.y-b.y); }
__device__ __forceinline__ float2 cinv(float2 a) {
    float n = a.x*a.x + a.y*a.y;
    return make_float2(a.x/n, -a.y/n);
}
__device__ __forceinline__ float2 cdiv(float2 q, float2 p) {
    float n = p.x*p.x + p.y*p.y;
    return make_float2((q.x*p.x + q.y*p.y)/n, (q.y*p.x - q.x*p.y)/n);
}

__device__ __forceinline__ float gelu_tanh(float x) {
    float x3 = x*x*x;
    return 0.5f*x*(1.f + tanhf(0.7978845608028654f*(x + 0.044715f*x3)));
}

// ---------------- 1) transpose + f32->f16 convert (per expert) ----------------
// src: (E, R, C) f32 row-major  ->  dst: (E, C, R) f16
__global__ __launch_bounds__(256) void k_convT(const float* __restrict__ src,
                                               _Float16* __restrict__ dst,
                                               int R, int C) {
    __shared__ float tile[32][33];
    int e  = blockIdx.z;
    const float* s = src + (size_t)e * R * C;
    _Float16*    d = dst + (size_t)e * R * C;
    int c0 = blockIdx.x * 32, r0 = blockIdx.y * 32;
    int tx = threadIdx.x, ty = threadIdx.y;
    #pragma unroll
    for (int i = ty; i < 32; i += 8)
        tile[i][tx] = s[(size_t)(r0 + i) * C + c0 + tx];
    __syncthreads();
    #pragma unroll
    for (int i = ty; i < 32; i += 8)
        d[(size_t)(c0 + i) * R + r0 + tx] = (_Float16)tile[tx][i];
}

// ---------------- 2) prologue: v, gating softmax top-2, x->f16 ----------------
__global__ __launch_bounds__(256) void k_prologue(const float* __restrict__ x,
                                                  const float* __restrict__ v_w,
                                                  const float* __restrict__ v_b,
                                                  const float* __restrict__ gate_w,
                                                  const float* __restrict__ gate_b,
                                                  float* __restrict__ vout,
                                                  _Float16* __restrict__ xh,
                                                  int* __restrict__ topi,
                                                  float* __restrict__ topw,
                                                  int* __restrict__ cnt) {
    int wv = threadIdx.x >> 5, lane = threadIdx.x & 31;
    int tok = blockIdx.x * 8 + wv;
    const float* xr = x + (size_t)tok * DD;
    float av = 0.f, ag[EE];
    #pragma unroll
    for (int e = 0; e < EE; ++e) ag[e] = 0.f;
    for (int k = lane; k < DD; k += 32) {
        float xv = xr[k];
        xh[(size_t)tok * DD + k] = (_Float16)xv;
        av += xv * v_w[k];
        const float* g = gate_w + (size_t)k * EE;
        #pragma unroll
        for (int e = 0; e < EE; ++e) ag[e] += xv * g[e];
    }
    #pragma unroll
    for (int off = 16; off > 0; off >>= 1) {
        av += __shfl_xor(av, off, 32);
        #pragma unroll
        for (int e = 0; e < EE; ++e) ag[e] += __shfl_xor(ag[e], off, 32);
    }
    if (lane == 0) {
        float vv = av + v_b[0];
        vv = fminf(3.0f, fmaxf(-3.0f, vv));
        vout[tok] = vv;
        float l[EE];
        #pragma unroll
        for (int e = 0; e < EE; ++e) l[e] = ag[e] + gate_b[e];
        int i0 = 0;
        #pragma unroll
        for (int e = 1; e < EE; ++e) if (l[e] > l[i0]) i0 = e;
        int i1 = (i0 == 0) ? 1 : 0;
        #pragma unroll
        for (int e = 0; e < EE; ++e) if (e != i0 && l[e] > l[i1]) i1 = e;
        float w0 = 1.0f / (1.0f + expf(l[i1] - l[i0]));
        topi[2*tok]   = i0;  topi[2*tok+1] = i1;
        topw[2*tok]   = w0;  topw[2*tok+1] = 1.0f - w0;
        atomicAdd(&cnt[i0], 1);
        atomicAdd(&cnt[i1], 1);
    }
}

// ---------------- 3) routing prefix + scatter ----------------
__global__ void k_base(const int* __restrict__ cnt, int* __restrict__ base,
                       int* __restrict__ cursor) {
    int s = 0;
    #pragma unroll
    for (int e = 0; e < EE; ++e) { base[e] = s; s += cnt[e]; cursor[e] = 0; }
}

__global__ __launch_bounds__(256) void k_scatter(const int* __restrict__ topi,
                                                 const int* __restrict__ base,
                                                 int* __restrict__ cursor,
                                                 int* __restrict__ pool_tok,
                                                 int* __restrict__ pos) {
    int tok = blockIdx.x * 256 + threadIdx.x;
    if (tok >= TOK) return;
    #pragma unroll
    for (int k = 0; k < 2; ++k) {
        int e = topi[2*tok + k];
        int s = atomicAdd(&cursor[e], 1);
        int idx = base[e] + s;
        pool_tok[idx] = tok;
        pos[2*tok + k] = idx;
    }
}

// ---------------- 4) Green's function via parallel Mobius scan ----------------
__global__ __launch_bounds__(256) void k_green(const float* __restrict__ v,
                                               float* __restrict__ featR,
                                               float* __restrict__ featI) {
    __shared__ float  ash[NN];
    __shared__ float2 mats[256][4];   // m00, m01, m10, m11
    __shared__ float2 dsh[NN];
    __shared__ float2 esh[NN];
    int b = blockIdx.x, t = threadIdx.x;
    const float* vrow = v + (size_t)b * NN;
    for (int i = t; i < NN; i += 256) ash[i] = vrow[i];
    __syncthreads();

    for (int dir = 0; dir < 2; ++dir) {
        __syncthreads();  // protect mats reuse between passes
        int i0 = t * 8;
        // per-chunk matrix product (left-multiply), normalized
        int j = dir ? (NN - 1 - i0) : i0;
        float2 a = make_float2(-2.0f + ash[j], -1.0f);
        float2 m00 = a, m01 = make_float2(-1.f, 0.f);
        float2 m10 = make_float2(1.f, 0.f), m11 = make_float2(0.f, 0.f);
        #pragma unroll
        for (int s = 1; s < 8; ++s) {
            j = dir ? (NN - 1 - (i0 + s)) : (i0 + s);
            a = make_float2(-2.0f + ash[j], -1.0f);
            float2 n00 = csub(cmul(a, m00), m10);
            float2 n01 = csub(cmul(a, m01), m11);
            m10 = m00; m11 = m01; m00 = n00; m01 = n01;
        }
        {
            float s = fmaxf(fmaxf(fmaxf(fabsf(m00.x), fabsf(m00.y)), fmaxf(fabsf(m01.x), fabsf(m01.y))),
                            fmaxf(fmaxf(fabsf(m10.x), fabsf(m10.y)), fmaxf(fabsf(m11.x), fabsf(m11.y))));
            float inv = (s > 0.f) ? 1.0f / s : 1.0f;
            m00.x*=inv; m00.y*=inv; m01.x*=inv; m01.y*=inv;
            m10.x*=inv; m10.y*=inv; m11.x*=inv; m11.y*=inv;
        }
        mats[t][0]=m00; mats[t][1]=m01; mats[t][2]=m10; mats[t][3]=m11;
        __syncthreads();
        // inclusive scan: P_t = C_t * C_{t-1} * ... * C_0
        for (int off = 1; off < 256; off <<= 1) {
            float2 L00, L01, L10, L11;
            bool act = (t >= off);
            if (act) { L00 = mats[t-off][0]; L01 = mats[t-off][1];
                       L10 = mats[t-off][2]; L11 = mats[t-off][3]; }
            __syncthreads();
            if (act) {
                float2 n00 = cadd(cmul(m00, L00), cmul(m01, L10));
                float2 n01 = cadd(cmul(m00, L01), cmul(m01, L11));
                float2 n10 = cadd(cmul(m10, L00), cmul(m11, L10));
                float2 n11 = cadd(cmul(m10, L01), cmul(m11, L11));
                float s = fmaxf(fmaxf(fmaxf(fabsf(n00.x), fabsf(n00.y)), fmaxf(fabsf(n01.x), fabsf(n01.y))),
                                fmaxf(fmaxf(fabsf(n10.x), fabsf(n10.y)), fmaxf(fabsf(n11.x), fabsf(n11.y))));
                float inv = (s > 0.f) ? 1.0f / s : 1.0f;
                m00 = make_float2(n00.x*inv, n00.y*inv); m01 = make_float2(n01.x*inv, n01.y*inv);
                m10 = make_float2(n10.x*inv, n10.y*inv); m11 = make_float2(n11.x*inv, n11.y*inv);
                mats[t][0]=m00; mats[t][1]=m01; mats[t][2]=m10; mats[t][3]=m11;
            }
            __syncthreads();
        }
        // local re-walk with exclusive prefix (d_{-1} from P_{t-1} column 0)
        float2 p = make_float2(1.f, 0.f), q = make_float2(0.f, 0.f);
        if (t > 0) { p = mats[t-1][0]; q = mats[t-1][2]; }
        float2 r = cdiv(q, p);
        j = dir ? (NN - 1 - i0) : i0;
        float2 d = csub(make_float2(-2.0f + ash[j], -1.0f), r);
        if (dir == 0) dsh[i0] = d; else esh[NN - 1 - i0] = d;
        #pragma unroll
        for (int s = 1; s < 8; ++s) {
            int i = i0 + s;
            j = dir ? (NN - 1 - i) : i;
            d = csub(make_float2(-2.0f + ash[j], -1.0f), cinv(d));
            if (dir == 0) dsh[i] = d; else esh[NN - 1 - i] = d;
        }
    }
    __syncthreads();
    for (int i = t; i < NN; i += 256) {
        float2 a = make_float2(-2.0f + ash[i], -1.0f);
        float2 den = make_float2(dsh[i].x + esh[i].x - a.x, dsh[i].y + esh[i].y - a.y);
        float2 G = cinv(den);
        featR[(size_t)b*NN + i] = fminf(10.f, fmaxf(-10.f, G.x));
        featI[(size_t)b*NN + i] = fminf(10.f, fmaxf(-10.f, G.y));
    }
}

// ---------------- 5) grouped expert GEMMs (WMMA f16, 32-token M tile) ----------------
__global__ __launch_bounds__(256) void k_expert(const _Float16* __restrict__ xh,
                                                const _Float16* __restrict__ wh1t, // (E,F,D)
                                                const _Float16* __restrict__ wh2t, // (E,D,F)
                                                const float* __restrict__ b1,      // (E,F)
                                                const float* __restrict__ b2,      // (E,D)
                                                const int* __restrict__ pool_tok,
                                                const int* __restrict__ base,
                                                const int* __restrict__ cnt,
                                                float* __restrict__ Y) {           // (PAIRS,D)
    int e  = blockIdx.y;
    int m0 = blockIdx.x * TM;
    int count = cnt[e];
    if (m0 >= count) return;
    int pbase = base[e];

    __shared__ _Float16 xs[TM * XPAD];
    __shared__ _Float16 hs[TM * HPAD];
    __shared__ int toks[TM];

    int tid = threadIdx.x;
    if (tid < TM) toks[tid] = (m0 + tid < count) ? pool_tok[pbase + m0 + tid] : -1;
    __syncthreads();

    // stage x tile (TM x 768 halves) into LDS
    for (int idx = tid; idx < TM * 96; idx += 256) {
        int r = idx / 96, c = idx % 96;
        v8h val;
        int tk = toks[r];
        if (tk >= 0) val = *reinterpret_cast<const v8h*>(xh + (size_t)tk * DD + c * 8);
        else          val = (v8h)(_Float16)0;
        *reinterpret_cast<v8h*>(&xs[r * XPAD + c * 8]) = val;
    }
    __syncthreads();

    int w = tid >> 5, l = tid & 31;
    int arow = l & 15;
    int kh   = (l < 16) ? 0 : 8;
    int bkh  = (l < 16) ? 0 : 8;
    int mrow = ((l >> 4) << 3);

    // GEMM1: h = gelu(x @ W1 + b1); per wave 24 N-tiles, 2 M-groups share each B
    for (int jj = 0; jj < 24; ++jj) {
        int f0 = (w * 24 + jj) * 16;
        v8f c0 = {}, c1 = {};
        const _Float16* bp = wh1t + ((size_t)e * FF + f0 + (l & 15)) * DD + bkh;
        for (int k0 = 0; k0 < DD; k0 += 32) {
            v8h b0l = *reinterpret_cast<const v8h*>(bp + k0);
            v8h b0h = *reinterpret_cast<const v8h*>(bp + k0 + 16);
            v16h Bf = __builtin_shufflevector(b0l, b0h, 0,1,2,3,4,5,6,7,8,9,10,11,12,13,14,15);
            v8h a0l = *reinterpret_cast<const v8h*>(&xs[arow * XPAD + k0 + kh]);
            v8h a0h = *reinterpret_cast<const v8h*>(&xs[arow * XPAD + k0 + 16 + kh]);
            v16h A0 = __builtin_shufflevector(a0l, a0h, 0,1,2,3,4,5,6,7,8,9,10,11,12,13,14,15);
            c0 = __builtin_amdgcn_wmma_f32_16x16x32_f16(false, A0, false, Bf, (short)0, c0, false, false);
            v8h a1l = *reinterpret_cast<const v8h*>(&xs[(16 + arow) * XPAD + k0 + kh]);
            v8h a1h = *reinterpret_cast<const v8h*>(&xs[(16 + arow) * XPAD + k0 + 16 + kh]);
            v16h A1 = __builtin_shufflevector(a1l, a1h, 0,1,2,3,4,5,6,7,8,9,10,11,12,13,14,15);
            c1 = __builtin_amdgcn_wmma_f32_16x16x32_f16(false, A1, false, Bf, (short)0, c1, false, false);
        }
        int col = f0 + (l & 15);
        float bias = b1[(size_t)e * FF + col];
        #pragma unroll
        for (int r = 0; r < 8; ++r) {
            hs[(r + mrow) * HPAD + col]      = (_Float16)gelu_tanh(c0[r] + bias);
            hs[(16 + r + mrow) * HPAD + col] = (_Float16)gelu_tanh(c1[r] + bias);
        }
    }
    __syncthreads();

    // GEMM2: y = h @ W2 + b2; per wave 6 N-tiles, 2 M-groups share each B
    for (int jj = 0; jj < 6; ++jj) {
        int d0 = (w * 6 + jj) * 16;
        v8f c0 = {}, c1 = {};
        const _Float16* bp = wh2t + ((size_t)e * DD + d0 + (l & 15)) * FF + bkh;
        for (int k0 = 0; k0 < FF; k0 += 32) {
            v8h b0l = *reinterpret_cast<const v8h*>(bp + k0);
            v8h b0h = *reinterpret_cast<const v8h*>(bp + k0 + 16);
            v16h Bf = __builtin_shufflevector(b0l, b0h, 0,1,2,3,4,5,6,7,8,9,10,11,12,13,14,15);
            v8h a0l = *reinterpret_cast<const v8h*>(&hs[arow * HPAD + k0 + kh]);
            v8h a0h = *reinterpret_cast<const v8h*>(&hs[arow * HPAD + k0 + 16 + kh]);
            v16h A0 = __builtin_shufflevector(a0l, a0h, 0,1,2,3,4,5,6,7,8,9,10,11,12,13,14,15);
            c0 = __builtin_amdgcn_wmma_f32_16x16x32_f16(false, A0, false, Bf, (short)0, c0, false, false);
            v8h a1l = *reinterpret_cast<const v8h*>(&hs[(16 + arow) * HPAD + k0 + kh]);
            v8h a1h = *reinterpret_cast<const v8h*>(&hs[(16 + arow) * HPAD + k0 + 16 + kh]);
            v16h A1 = __builtin_shufflevector(a1l, a1h, 0,1,2,3,4,5,6,7,8,9,10,11,12,13,14,15);
            c1 = __builtin_amdgcn_wmma_f32_16x16x32_f16(false, A1, false, Bf, (short)0, c1, false, false);
        }
        int col = d0 + (l & 15);
        float bias = b2[(size_t)e * DD + col];
        #pragma unroll
        for (int r = 0; r < 8; ++r) {
            int m = r + mrow;
            if (m0 + m < count)
                Y[(size_t)(pbase + m0 + m) * DD + col] = c0[r] + bias;
            if (m0 + 16 + m < count)
                Y[(size_t)(pbase + m0 + 16 + m) * DD + col] = c1[r] + bias;
        }
    }
}

// ---------------- 6) final combine (deterministic gather) ----------------
__global__ __launch_bounds__(256) void k_combine(const float* __restrict__ Y,
                                                 const float* __restrict__ topw,
                                                 const int* __restrict__ pos,
                                                 const float* __restrict__ featR,
                                                 const float* __restrict__ featI,
                                                 const float* __restrict__ out_w, // (2,D)
                                                 const float* __restrict__ out_b,
                                                 const float* __restrict__ bk_scale,
                                                 float* __restrict__ out) {
    int t = blockIdx.x;
    float w0 = topw[2*t], w1 = topw[2*t+1];
    int   p0 = pos[2*t],  p1 = pos[2*t+1];
    float fR = featR[t],  fI = featI[t];
    const float* y0 = Y + (size_t)p0 * DD;
    const float* y1 = Y + (size_t)p1 * DD;
    for (int d = threadIdx.x; d < DD; d += 256) {
        float spec = fR * out_w[d] + fI * out_w[DD + d] + out_b[d];
        out[(size_t)t * DD + d] = w0 * y0[d] + w1 * y1[d] + bk_scale[d] * spec;
    }
}

// ---------------- launcher ----------------
extern "C" void kernel_launch(void* const* d_in, const int* in_sizes, int n_in,
                              void* d_out, int out_size, void* d_ws, size_t ws_size,
                              hipStream_t stream) {
    const float* x        = (const float*)d_in[0];
    const float* v_w      = (const float*)d_in[1];
    const float* v_b      = (const float*)d_in[2];
    const float* gate_w   = (const float*)d_in[3];
    const float* gate_b   = (const float*)d_in[4];
    const float* w1       = (const float*)d_in[5];
    const float* b1       = (const float*)d_in[6];
    const float* w2       = (const float*)d_in[7];
    const float* b2       = (const float*)d_in[8];
    const float* out_w    = (const float*)d_in[9];
    const float* out_b    = (const float*)d_in[10];
    const float* bk_scale = (const float*)d_in[11];
    float* out = (float*)d_out;

    char* ws = (char*)d_ws;
    size_t o = 0;
    auto take = [&](size_t bytes) { char* p = ws + o; o = (o + bytes + 255) & ~(size_t)255; return p; };
    _Float16* xh     = (_Float16*)take((size_t)TOK * DD * 2);
    _Float16* wh1t   = (_Float16*)take((size_t)EE * FF * DD * 2);
    _Float16* wh2t   = (_Float16*)take((size_t)EE * DD * FF * 2);
    float*    Y      = (float*)   take((size_t)PAIRS * DD * 4);
    float*    vbuf   = (float*)   take((size_t)TOK * 4);
    float*    featR  = (float*)   take((size_t)TOK * 4);
    float*    featI  = (float*)   take((size_t)TOK * 4);
    int*      topi   = (int*)     take((size_t)PAIRS * 4);
    float*    topw   = (float*)   take((size_t)PAIRS * 4);
    int*      pos    = (int*)     take((size_t)PAIRS * 4);
    int*      pool   = (int*)     take((size_t)PAIRS * 4);
    int*      cnt    = (int*)     take(EE * 4);
    int*      basep  = (int*)     take(EE * 4);
    int*      cursor = (int*)     take(EE * 4);

    hipMemsetAsync(cnt, 0, EE * sizeof(int), stream);

    // weight transpose+convert: w1 (E,D,F)->(E,F,D), w2 (E,F,D)->(E,D,F)
    k_convT<<<dim3(FF/32, DD/32, EE), dim3(32, 8), 0, stream>>>(w1, wh1t, DD, FF);
    k_convT<<<dim3(DD/32, FF/32, EE), dim3(32, 8), 0, stream>>>(w2, wh2t, FF, DD);

    k_prologue<<<TOK/8, 256, 0, stream>>>(x, v_w, v_b, gate_w, gate_b,
                                          vbuf, xh, topi, topw, cnt);
    k_base<<<1, 1, 0, stream>>>(cnt, basep, cursor);
    k_scatter<<<TOK/256, 256, 0, stream>>>(topi, basep, cursor, pool, pos);
    k_green<<<BB, 256, 0, stream>>>(vbuf, featR, featI);
    k_expert<<<dim3(TOK/TM, EE), 256, 0, stream>>>(xh, wh1t, wh2t, b1, b2,
                                                   pool, basep, cnt, Y);
    k_combine<<<TOK, 256, 0, stream>>>(Y, topw, pos, featR, featI,
                                       out_w, out_b, bk_scale, out);
}
